// LSTMCell_89515708383954
// MI455X (gfx1250) — matrix operand order repaired and checked by reading, not compiled, not run on hardware
//
#include <hip/hip_runtime.h>

#define LSTM_BATCH  2048
#define LSTM_INPUT  4096
#define LSTM_HIDDEN 4096

// WMMA operand types (gfx1250 / CDNA5, wave32)
typedef __attribute__((ext_vector_type(16))) __bf16 bf16x16;
typedef __attribute__((ext_vector_type(8)))  __bf16 bf16x8;
typedef __attribute__((ext_vector_type(8)))  float  v8f;

union FragU { bf16x16 v; bf16x8 h[2]; };

// Split one fp32 into truncated-bf16 hi and RNE-bf16 lo of the residual,
// pack 4 elements (one float4) into two bf16 LDS rows (hi / lo).
__device__ __forceinline__ void cvt_split_store(unsigned short* dhi,
                                                unsigned short* dlo,
                                                float4 v) {
    float f[4] = {v.x, v.y, v.z, v.w};
    unsigned hw[4], lw[4];
#pragma unroll
    for (int i = 0; i < 4; ++i) {
        union { float f; unsigned u; } a; a.f = f[i];
        unsigned hbits = a.u & 0xFFFF0000u;          // truncate -> bf16 hi
        union { float f; unsigned u; } hf; hf.u = hbits;
        float r = f[i] - hf.f;                       // exact residual
        union { float f; unsigned u; } rb; rb.f = r;
        unsigned lbits = rb.u + 0x7FFFu + ((rb.u >> 16) & 1u); // RNE -> bf16 lo
        hw[i] = hbits >> 16;
        lw[i] = lbits >> 16;
    }
    uint2 hp; hp.x = hw[0] | (hw[1] << 16); hp.y = hw[2] | (hw[3] << 16);
    uint2 lp; lp.x = lw[0] | (lw[1] << 16); lp.y = lw[2] | (lw[3] << 16);
    *reinterpret_cast<uint2*>(dhi) = hp;
    *reinterpret_cast<uint2*>(dlo) = lp;
}

__device__ __forceinline__ float sigmoid_fast(float x) {
    return 1.0f / (1.0f + __expf(-x));
}
__device__ __forceinline__ float tanh_fast(float x) {
    return 2.0f / (1.0f + __expf(-2.0f * x)) - 1.0f;
}

// Fused LSTM cell (h_prev = c_prev = 0):
//   G = x @ W_ih^T for gates {i, g, o} only (f gate multiplies c_prev==0)
//   cell   = sigmoid(i) * tanh(g)
//   hidden = sigmoid(o) * tanh(cell)
// Block tile: 128 (batch) x 64 (hidden), 8 waves; each wave: 64x16 per gate,
// all 3 gates resident so the epilogue is wave-local.
__global__ __launch_bounds__(256)
void lstm_cell_wmma_bf16x3(const float* __restrict__ x,
                           const float* __restrict__ w,
                           float* __restrict__ out) {
    // K-stride padded 32 -> 40 elems (80B = 5*16B) to avoid LDS bank conflicts
    __shared__ unsigned short Ahi[128][40];
    __shared__ unsigned short Alo[128][40];
    __shared__ unsigned short Bhi[3][64][40];
    __shared__ unsigned short Blo[3][64][40];

    const int tid    = threadIdx.x;
    const int lane   = tid & 31;
    const int wv     = tid >> 5;     // 0..7
    const int waveM  = wv & 1;       // 2 waves along M -> 64 rows each
    const int waveH  = wv >> 1;      // 4 waves along H -> 16 cols each
    const int lr     = lane & 15;
    const int hiHalf = lane >> 4;    // 0 or 1

    const int blockH0 = blockIdx.x * 64;
    const int blockM0 = blockIdx.y * 128;

    // cooperative global->LDS mapping
    const int arow = tid >> 1;          // 0..127, 2 threads/row
    const int acol = (tid & 1) * 16;    // 16 floats each
    const int brow = tid >> 2;          // 0..63, 4 threads/row
    const int bcol = (tid & 3) * 8;     // 8 floats each

    // W_ih row bases for gates i, g, o (PyTorch gate order i,f,g,o; f skipped)
    const int wrow0[3] = {0 * LSTM_HIDDEN, 2 * LSTM_HIDDEN, 3 * LSTM_HIDDEN};

    const float* xrow = x + (size_t)(blockM0 + arow) * LSTM_INPUT + acol;
    const float* wrow[3];
#pragma unroll
    for (int q = 0; q < 3; ++q)
        wrow[q] = w + (size_t)(wrow0[q] + blockH0 + brow) * LSTM_INPUT + bcol;

    v8f acc[3][4];
#pragma unroll
    for (int g = 0; g < 3; ++g)
#pragma unroll
        for (int mi = 0; mi < 4; ++mi)
            acc[g][mi] = v8f{0.f, 0.f, 0.f, 0.f, 0.f, 0.f, 0.f, 0.f};

#pragma unroll 1
    for (int k0 = 0; k0 < LSTM_INPUT; k0 += 32) {
        // ---- global loads (fp32, 128B/row-chunk, fully coalesced) ----
        float4 av[4];
#pragma unroll
        for (int j = 0; j < 4; ++j)
            av[j] = reinterpret_cast<const float4*>(xrow + k0)[j];
        float4 bv[3][2];
#pragma unroll
        for (int q = 0; q < 3; ++q)
#pragma unroll
            for (int j = 0; j < 2; ++j)
                bv[q][j] = reinterpret_cast<const float4*>(wrow[q] + k0)[j];

        if (k0 + 32 < LSTM_INPUT) {   // global_prefetch_b8 next K tile
            __builtin_prefetch(xrow + k0 + 32, 0, 3);
#pragma unroll
            for (int q = 0; q < 3; ++q)
                __builtin_prefetch(wrow[q] + k0 + 32, 0, 3);
        }

        __syncthreads();  // previous iteration's frags consumed

        // ---- split fp32 -> bf16 hi/lo, store to LDS ----
#pragma unroll
        for (int j = 0; j < 4; ++j)
            cvt_split_store(&Ahi[arow][acol + 4 * j],
                            &Alo[arow][acol + 4 * j], av[j]);
#pragma unroll
        for (int q = 0; q < 3; ++q)
#pragma unroll
            for (int j = 0; j < 2; ++j)
                cvt_split_store(&Bhi[q][brow][bcol + 4 * j],
                                &Blo[q][brow][bcol + 4 * j], bv[q][j]);

        __syncthreads();

        // ---- A fragments (16-bit A 16x32 layout):
        // lanes 0-15: K 0-7 & 16-23 ; lanes 16-31: K 8-15 & 24-31
        const int ka = hiHalf * 8;
        FragU fahi[4], falo[4];
#pragma unroll
        for (int mi = 0; mi < 4; ++mi) {
            const unsigned short* ah = &Ahi[waveM * 64 + mi * 16 + lr][0];
            const unsigned short* al = &Alo[waveM * 64 + mi * 16 + lr][0];
            fahi[mi].h[0] = *reinterpret_cast<const bf16x8*>(ah + ka);
            fahi[mi].h[1] = *reinterpret_cast<const bf16x8*>(ah + 16 + ka);
            falo[mi].h[0] = *reinterpret_cast<const bf16x8*>(al + ka);
            falo[mi].h[1] = *reinterpret_cast<const bf16x8*>(al + 16 + ka);
        }

        // ---- B fragments (32x16): lanes 0-15: K 0-7 & 16-23, lanes 16-31: K 8-15 & 24-31
        const int kb = hiHalf * 8;
#pragma unroll
        for (int g = 0; g < 3; ++g) {
            const unsigned short* bh = &Bhi[g][waveH * 16 + lr][0];
            const unsigned short* bl = &Blo[g][waveH * 16 + lr][0];
            FragU fbhi, fblo;
            fbhi.h[0] = *reinterpret_cast<const bf16x8*>(bh + kb);
            fbhi.h[1] = *reinterpret_cast<const bf16x8*>(bh + kb + 16);
            fblo.h[0] = *reinterpret_cast<const bf16x8*>(bl + kb);
            fblo.h[1] = *reinterpret_cast<const bf16x8*>(bl + kb + 16);

#pragma unroll
            for (int mi = 0; mi < 4; ++mi) {
                // bf16x3 split product: hi*hi + hi*lo + lo*hi  (~fp32 accuracy)
                acc[g][mi] = __builtin_amdgcn_wmma_f32_16x16x32_bf16(
                    false, fahi[mi].v, false, fbhi.v, (short)0, acc[g][mi], false, false);
                acc[g][mi] = __builtin_amdgcn_wmma_f32_16x16x32_bf16(
                    false, fahi[mi].v, false, fblo.v, (short)0, acc[g][mi], false, false);
                acc[g][mi] = __builtin_amdgcn_wmma_f32_16x16x32_bf16(
                    false, falo[mi].v, false, fbhi.v, (short)0, acc[g][mi], false, false);
            }
        }
    }

    // ---- fused LSTM epilogue (c_prev = 0) ----
    // C/D layout: VGPR r, lanes 0-15 -> M=r, N=lane; lanes 16-31 -> M=r+8
    float* hiddenp = out;
    float* cellp   = out + (size_t)LSTM_BATCH * LSTM_HIDDEN;
    const int h = blockH0 + waveH * 16 + lr;
#pragma unroll
    for (int mi = 0; mi < 4; ++mi) {
#pragma unroll
        for (int r = 0; r < 8; ++r) {
            const int m = blockM0 + waveM * 64 + mi * 16 + r + hiHalf * 8;
            const float ig = sigmoid_fast(acc[0][mi][r]);
            const float gg = tanh_fast(acc[1][mi][r]);
            const float og = sigmoid_fast(acc[2][mi][r]);
            const float c  = ig * gg;
            const float hd = og * tanh_fast(c);
            hiddenp[(size_t)m * LSTM_HIDDEN + h] = hd;
            cellp[(size_t)m * LSTM_HIDDEN + h]   = c;
        }
    }
}

extern "C" void kernel_launch(void* const* d_in, const int* in_sizes, int n_in,
                              void* d_out, int out_size, void* d_ws, size_t ws_size,
                              hipStream_t stream) {
    const float* x    = (const float*)d_in[0];
    const float* w_ih = (const float*)d_in[1];
    // d_in[2] = weight_hh is mathematically dead: h_prev == 0.
    (void)in_sizes; (void)n_in; (void)d_ws; (void)ws_size; (void)out_size;
    float* out = (float*)d_out;

    dim3 grid(LSTM_HIDDEN / 64, LSTM_BATCH / 128);  // 64 x 16 = 1024 blocks
    dim3 block(256);                                 // 8 waves (wave32)
    lstm_cell_wmma_bf16x3<<<grid, block, 0, stream>>>(x, w_ih, out);
}
